// GATEncoder_18915035972102
// MI455X (gfx1250) — matrix operand order repaired and checked by reading, not compiled, not run on hardware
//
#include <hip/hip_runtime.h>
#include <hip/hip_bf16.h>

// ---- problem sizes (match reference) ----
#define NN   50000     // nodes
#define EE   800000    // edges before self loops
#define EP   (EE + NN) // edges incl self loops = 850000
#define IN_D 128
#define H1   8
#define C1   32
#define D1   (H1 * C1) // 256
#define OUTD 64

typedef __attribute__((ext_vector_type(16))) _Float16 v16h;
typedef __attribute__((ext_vector_type(8)))  _Float16 v8h;
typedef __attribute__((ext_vector_type(8)))  float    v8f;

static inline int cdiv(long long a, long long b) { return (int)((a + b - 1) / b); }

// -------------------------------------------------------------------------
// WMMA GEMM:  C[M, NCOLS] = A[M, K] (f16, row major) * Bt[NCOLS, K]^T (f16)
// One wave computes a 16x64 strip: the A fragment is loaded once per
// 32-wide K chunk and reused across 4 v_wmma_f32_16x16x32_f16 issues
// (4 f32 accumulators), improving the load:WMMA ratio from 4:1 to 2.5:1
// and cutting A-matrix global traffic 4x vs a 16x16-per-wave tiling.
// Fragment layouts per CDNA5 ISA 7.12.2 (A: lanes 0-15 hold K {0..7,16..23},
// lanes 16-31 K {8..15,24..31}; B: lane = column, same K split as A).
// -------------------------------------------------------------------------
template<int K, int NCOLS>
__global__ __launch_bounds__(256) void wmma_gemm4(
    const _Float16* __restrict__ A, const _Float16* __restrict__ Bt,
    float* __restrict__ C, int Mtiles) {
  constexpr int NG = NCOLS / 64;             // 64-column groups per row strip
  const int wave = (blockIdx.x * blockDim.x + threadIdx.x) >> 5;
  const int lane = threadIdx.x & 31;
  const int mt = wave / NG;
  const int ng = wave % NG;
  if (mt >= Mtiles) return;                  // wave-uniform: EXEC stays all-1s

  const int hi   = lane >> 4;                // 0: lanes 0-15, 1: lanes 16-31
  const int mrow = mt * 16 + (lane & 15);    // A fragment: row per lane
  const int aoff = hi ? 8  : 0;              // A K-group offset
  const int boff = hi ? 8  : 0;              // B K offset for upper lanes
  const int ncol0 = ng * 64 + (lane & 15);   // B fragment: column per lane

  v8f acc[4] = {};
#pragma unroll
  for (int kc = 0; kc < K; kc += 32) {
    const _Float16* ap = A + (long long)mrow * K + kc + aoff;
    v8h alo = *(const v8h*)(ap);
    v8h ahi = *(const v8h*)(ap + 16);
    v16h a;
#pragma unroll
    for (int i = 0; i < 8; ++i) { a[i] = alo[i]; a[i + 8] = ahi[i]; }

#pragma unroll
    for (int t = 0; t < 4; ++t) {
      const _Float16* bp = Bt + (long long)(ncol0 + t * 16) * K + kc + boff;
      v8h blo = *(const v8h*)(bp);
      v8h bhi = *(const v8h*)(bp + 16);
      v16h b;
#pragma unroll
      for (int i = 0; i < 8; ++i) { b[i] = blo[i]; b[i + 8] = bhi[i]; }
      acc[t] = __builtin_amdgcn_wmma_f32_16x16x32_f16(false, a, false, b,
                                                      (short)0, acc[t], false, false);
    }
  }
  // C/D layout: lane -> column (lane&15), VGPR j -> row j + 8*(lane>=16)
#pragma unroll
  for (int t = 0; t < 4; ++t) {
#pragma unroll
    for (int j = 0; j < 8; ++j) {
      const int m = mt * 16 + j + (hi ? 8 : 0);
      C[(long long)m * NCOLS + ng * 64 + t * 16 + (lane & 15)] = acc[t][j];
    }
  }
}

// ---- f32 -> f16 elementwise ----
__global__ void f32_to_f16(const float* __restrict__ in, _Float16* __restrict__ out, int n) {
  int i = blockIdx.x * blockDim.x + threadIdx.x;
  if (i < n) out[i] = (_Float16)in[i];
}

// ---- f32 [K,Nc] -> f16 transposed [Nc,K] ----
__global__ void transpose_f16(const float* __restrict__ in, _Float16* __restrict__ out,
                              int K, int Nc) {
  int i = blockIdx.x * blockDim.x + threadIdx.x;
  if (i >= K * Nc) return;
  int k = i / Nc, n = i % Nc;
  out[n * K + k] = (_Float16)in[i];
}

__global__ void zero_u32(unsigned* __restrict__ p, int n) {
  int i = blockIdx.x * blockDim.x + threadIdx.x;
  if (i < n) p[i] = 0u;
}

// ---- per-(node,head) attention logits: one wave32 per (node,head) ----
__global__ void attn_logits(const float* __restrict__ h, const float* __restrict__ a_src,
                            const float* __restrict__ a_dst, float* __restrict__ al,
                            float* __restrict__ ar, int nNodes, int heads, int cdim) {
  int warp = (blockIdx.x * blockDim.x + threadIdx.x) >> 5;
  int lane = threadIdx.x & 31;
  if (warp >= nNodes * heads) return;
  int node = warp / heads, head = warp % heads;
  float sa = 0.f, sb = 0.f;
  for (int c = lane; c < cdim; c += 32) {
    float v = h[((long long)node * heads + head) * cdim + c];
    sa += v * a_src[head * cdim + c];
    sb += v * a_dst[head * cdim + c];
  }
  for (int off = 16; off; off >>= 1) {
    sa += __shfl_down(sa, off, 32);
    sb += __shfl_down(sb, off, 32);
  }
  if (lane == 0) { al[warp] = sa; ar[warp] = sb; }
}

// ---- order-preserving float<->uint encoding for atomicMax ----
__device__ __forceinline__ unsigned enc_f(float f) {
  unsigned u = __float_as_uint(f);
  return (u & 0x80000000u) ? ~u : (u | 0x80000000u);
}
__device__ __forceinline__ float dec_f(unsigned k) {
  return __uint_as_float((k & 0x80000000u) ? (k & 0x7fffffffu) : ~k);
}

__device__ __forceinline__ void edge_sd(const long long* __restrict__ ei, int idx,
                                        int& s, int& d) {
  if (idx < EE) { s = (int)ei[idx]; d = (int)ei[EE + idx]; }
  else          { s = d = idx - EE; }
}

// ---- pass 1: logits + segment max (atomicMax on encoded floats) ----
__global__ void edge_max(const long long* __restrict__ ei,
                         const float* __restrict__ al, const float* __restrict__ ar,
                         float* __restrict__ ebuf, unsigned* __restrict__ menc, int heads) {
  int idx = blockIdx.x * blockDim.x + threadIdx.x;
  if (idx >= EP) return;
  int s, d; edge_sd(ei, idx, s, d);
  for (int h = 0; h < heads; ++h) {
    float e = al[s * heads + h] + ar[d * heads + h];
    e = (e > 0.f) ? e : 0.2f * e;                 // LeakyReLU(0.2)
    ebuf[(long long)idx * heads + h] = e;
    atomicMax(&menc[d * heads + h], enc_f(e));
  }
}

// ---- pass 2: p = exp(e - max); segment sum ----
__global__ void edge_sum(const long long* __restrict__ ei,
                         float* __restrict__ ebuf, const unsigned* __restrict__ menc,
                         float* __restrict__ ssum, int heads) {
  int idx = blockIdx.x * blockDim.x + threadIdx.x;
  if (idx >= EP) return;
  int s, d; edge_sd(ei, idx, s, d);
  for (int h = 0; h < heads; ++h) {
    float m = dec_f(menc[d * heads + h]);
    float p = __expf(ebuf[(long long)idx * heads + h] - m);
    ebuf[(long long)idx * heads + h] = p;
    atomicAdd(&ssum[d * heads + h], p);
  }
}

// ---- pass 3: out[dst] += h[src] * alpha  (one wave32 per edge) ----
__global__ void edge_agg(const long long* __restrict__ ei,
                         const float* __restrict__ hfeat, const float* __restrict__ p,
                         const float* __restrict__ ssum, float* __restrict__ out,
                         int heads, int cdim) {
  int wave = (blockIdx.x * blockDim.x + threadIdx.x) >> 5;
  int lane = threadIdx.x & 31;
  if (wave >= EP) return;
  int s, d; edge_sd(ei, wave, s, d);
  const int D = heads * cdim;
  for (int ch = lane, h = 0; ch < D; ch += 32, h = (ch / cdim)) {
    // alpha is uniform across the wave for a given head iteration
    float alpha = p[(long long)wave * heads + h] *
                  __frcp_rn(ssum[d * heads + h] + 1e-16f);
    atomicAdd(&out[(long long)d * D + ch], hfeat[(long long)s * D + ch] * alpha);
  }
}

// ---- layer-1 epilogue: bias + ELU, emit f16 for the next GEMM ----
__global__ void finalize1(const float* __restrict__ out1, const float* __restrict__ b1,
                          _Float16* __restrict__ hact16, int total) {
  int i = blockIdx.x * blockDim.x + threadIdx.x;
  if (i >= total) return;
  float v = out1[i] + b1[i & (D1 - 1)];
  v = (v > 0.f) ? v : (__expf(v) - 1.f);          // ELU
  hact16[i] = (_Float16)v;
}

// ---- final bias add in place ----
__global__ void bias_add(float* __restrict__ out, const float* __restrict__ b,
                         int total, int D) {
  int i = blockIdx.x * blockDim.x + threadIdx.x;
  if (i >= total) return;
  out[i] += b[i % D];
}

extern "C" void kernel_launch(void* const* d_in, const int* in_sizes, int n_in,
                              void* d_out, int out_size, void* d_ws, size_t ws_size,
                              hipStream_t stream) {
  const float*     x      = (const float*)d_in[0];
  const long long* ei     = (const long long*)d_in[1];   // int64 [2,E]
  const float*     W1     = (const float*)d_in[2];
  const float*     a_src1 = (const float*)d_in[3];
  const float*     a_dst1 = (const float*)d_in[4];
  const float*     b1     = (const float*)d_in[5];
  const float*     W2     = (const float*)d_in[6];
  const float*     a_src2 = (const float*)d_in[7];
  const float*     a_dst2 = (const float*)d_in[8];
  const float*     b2     = (const float*)d_in[9];
  float*           out    = (float*)d_out;               // [N, 64]

  // ---- bump allocator over d_ws (~150 MB total, buffers overlaid) ----
  char* base = (char*)d_ws;
  size_t off = 0;
  auto alloc = [&](size_t bytes) -> char* {
    off = (off + 255) & ~(size_t)255;
    char* p = base + off;
    off += bytes;
    return p;
  };
  _Float16* w1t    = (_Float16*)alloc((size_t)IN_D * D1 * 2);     // [256,128] f16
  _Float16* w2t    = (_Float16*)alloc((size_t)D1 * OUTD * 2);     // [64,256]  f16
  char*     x16_g2 = alloc((size_t)NN * IN_D * 2);                // x16, later g2 (same bytes)
  char*     h1_ha  = alloc((size_t)NN * D1 * 4);                  // h1 f32, later hact f16
  float*    ebuf   = (float*)alloc((size_t)EP * H1 * 4);          // e/p layer1; layer2 reuses
  unsigned* menc   = (unsigned*)alloc((size_t)NN * H1 * 4);
  float*    ssum   = (float*)alloc((size_t)NN * H1 * 4);
  float*    al     = (float*)alloc((size_t)NN * H1 * 4);
  float*    ar     = (float*)alloc((size_t)NN * H1 * 4);
  float*    out1   = (float*)alloc((size_t)NN * D1 * 4);

  _Float16* x16    = (_Float16*)x16_g2;
  float*    g2     = (float*)x16_g2;        // after GEMM1 input (x16) consumed
  float*    h1     = (float*)h1_ha;
  _Float16* hact16 = (_Float16*)h1_ha;      // after layer-1 aggregation done

  const int Mtiles = NN / 16;               // 3125, exact

  // ---- precision conversion / transposes ----
  f32_to_f16<<<cdiv((long long)NN * IN_D, 256), 256, 0, stream>>>(x, x16, NN * IN_D);
  transpose_f16<<<cdiv(IN_D * D1, 256), 256, 0, stream>>>(W1, w1t, IN_D, D1);
  transpose_f16<<<cdiv(D1 * OUTD, 256), 256, 0, stream>>>(W2, w2t, D1, OUTD);

  // ---- layer 1: GEMM (WMMA), logits, softmax, aggregate ----
  {
    int waves = Mtiles * (D1 / 64);                       // 12500
    wmma_gemm4<IN_D, D1><<<cdiv(waves, 8), 256, 0, stream>>>(x16, w1t, h1, Mtiles);
  }
  attn_logits<<<cdiv((long long)NN * H1, 8), 256, 0, stream>>>(h1, a_src1, a_dst1,
                                                               al, ar, NN, H1, C1);
  zero_u32<<<cdiv(NN * H1, 256), 256, 0, stream>>>(menc, NN * H1);
  zero_u32<<<cdiv(NN * H1, 256), 256, 0, stream>>>((unsigned*)ssum, NN * H1);
  edge_max<<<cdiv(EP, 256), 256, 0, stream>>>(ei, al, ar, ebuf, menc, H1);
  edge_sum<<<cdiv(EP, 256), 256, 0, stream>>>(ei, ebuf, menc, ssum, H1);
  zero_u32<<<cdiv((long long)NN * D1, 256), 256, 0, stream>>>((unsigned*)out1, NN * D1);
  edge_agg<<<cdiv((long long)EP * 32, 256), 256, 0, stream>>>(ei, h1, ebuf, ssum,
                                                              out1, H1, C1);
  finalize1<<<cdiv((long long)NN * D1, 256), 256, 0, stream>>>(out1, b1, hact16, NN * D1);

  // ---- layer 2: GEMM (WMMA), logits, softmax, aggregate into d_out ----
  {
    int waves = Mtiles * (OUTD / 64);                     // 3125
    wmma_gemm4<D1, OUTD><<<cdiv(waves, 8), 256, 0, stream>>>(hact16, w2t, g2, Mtiles);
  }
  attn_logits<<<cdiv(NN, 8), 256, 0, stream>>>(g2, a_src2, a_dst2, al, ar, NN, 1, OUTD);
  zero_u32<<<cdiv(NN, 256), 256, 0, stream>>>(menc, NN);
  zero_u32<<<cdiv(NN, 256), 256, 0, stream>>>((unsigned*)ssum, NN);
  edge_max<<<cdiv(EP, 256), 256, 0, stream>>>(ei, al, ar, ebuf, menc, 1);
  edge_sum<<<cdiv(EP, 256), 256, 0, stream>>>(ei, ebuf, menc, ssum, 1);
  zero_u32<<<cdiv((long long)NN * OUTD, 256), 256, 0, stream>>>((unsigned*)out, NN * OUTD);
  edge_agg<<<cdiv((long long)EP * 32, 256), 256, 0, stream>>>(ei, g2, ebuf, ssum,
                                                              out, 1, OUTD);
  bias_add<<<cdiv(NN * OUTD, 256), 256, 0, stream>>>(out, b2, NN * OUTD, OUTD);
}
